// AttentionLayer_41420664602657
// MI455X (gfx1250) — matrix operand order repaired and checked by reading, not compiled, not run on hardware
//
#include <hip/hip_runtime.h>

typedef __attribute__((ext_vector_type(16))) _Float16 v16h;
typedef __attribute__((ext_vector_type(8)))  _Float16 v8h;
typedef __attribute__((ext_vector_type(8)))  float    v8f;

#define B_ 4
#define L_ 2048
#define S_ 2048
#define D_ 512
#define H_ 8
#define E_ 64

static __device__ __forceinline__ v8f wmma16(v16h a, v16h b, v8f c) {
  // v_wmma_f32_16x16x32_f16
  return __builtin_amdgcn_wmma_f32_16x16x32_f16(false, a, false, b, (short)0, c, false, false);
}

// A-fragment (16x32 f16) per-lane data: lane (m = lane&15, hi = lane>>4) holds
// row m, K = {base..base+7, base+16..base+23}, base = 8*hi. `p` points at K=base.
static __device__ __forceinline__ v16h load_a_f16(const _Float16* p) {
  union { v16h v; v8h h[2]; } u;
  u.h[0] = *(const v8h*)(p);
  u.h[1] = *(const v8h*)(p + 16);
  return u.v;
}

// Same A-fragment pattern but converting from f32 source
static __device__ __forceinline__ v16h load_a_f32cvt(const float* p) {
  v16h r;
#pragma unroll
  for (int i = 0; i < 8; ++i) {
    r[i]     = (_Float16)p[i];
    r[i + 8] = (_Float16)p[i + 16];
  }
  return r;
}

// B-fragment (32x16 f16): lane (n = lane&15, hi = lane>>4) holds column n,
// K = {base..base+7, base+16..base+23}, base = 8*hi, from a [n][k] (column-major-of-B) buffer.
static __device__ __forceinline__ v16h load_b_f16(const _Float16* p) {
  union { v16h v; v8h h[2]; } u;
  u.h[0] = *(const v8h*)(p);
  u.h[1] = *(const v8h*)(p + 16);
  return u.v;
}

// ---------------------------------------------------------------------------
// Kernel 0: convert 4 weight matrices f32 [k][n] -> f16 transposed [n][k]
// ---------------------------------------------------------------------------
__global__ void wconv_kernel(const float* __restrict__ Wq, const float* __restrict__ Wk,
                             const float* __restrict__ Wv, const float* __restrict__ Wo,
                             _Float16* __restrict__ Wt) {
  int idx = blockIdx.x * 256 + threadIdx.x;       // 4 * 512 * 512 threads
  int w   = idx >> 18;
  int rem = idx & 262143;
  int n = rem >> 9;
  int k = rem & 511;
  const float* W = (w == 0) ? Wq : (w == 1) ? Wk : (w == 2) ? Wv : Wo;
  Wt[(size_t)w * (D_ * D_) + (size_t)n * D_ + k] = (_Float16)W[(size_t)k * D_ + n];
}

// ---------------------------------------------------------------------------
// Kernel 1: fused QKV projection. z=0:Q (pre-scaled by 1/sqrt(E)), z=1:K,
// z=2:V (transposed store [B,H,E,S]).
// Wave computes a 16(M) x 64(N) tile with 4 f32 accumulators, K-loop step 32.
// ---------------------------------------------------------------------------
__global__ void __launch_bounds__(256)
qkv_proj_kernel(const float* __restrict__ Xq, const float* __restrict__ Xk,
                const float* __restrict__ Xv, const _Float16* __restrict__ Wt,
                const float* __restrict__ bq, const float* __restrict__ bk,
                const float* __restrict__ bv,
                _Float16* __restrict__ Qh, _Float16* __restrict__ Kh,
                _Float16* __restrict__ Vt) {
  const int z = blockIdx.z;
  const float* X = (z == 0) ? Xq : (z == 1) ? Xk : Xv;
  const _Float16* W = Wt + (size_t)z * (D_ * D_);
  const float* bias = (z == 0) ? bq : (z == 1) ? bk : bv;

  const int lane = threadIdx.x & 31;
  const int wave = threadIdx.x >> 5;
  const int lm = lane & 15;
  const int hi = lane >> 4;
  const int m0 = blockIdx.x * 128 + wave * 16;
  const int n0 = blockIdx.y * 64;

  v8f acc[4];
#pragma unroll
  for (int j = 0; j < 4; ++j)
#pragma unroll
    for (int i = 0; i < 8; ++i) acc[j][i] = 0.0f;

  const float* arow = X + (size_t)(m0 + lm) * D_;
  for (int kb = 0; kb < D_; kb += 32) {
    v16h a = load_a_f32cvt(arow + kb + hi * 8);
#pragma unroll
    for (int j = 0; j < 4; ++j) {
      v16h b = load_b_f16(W + (size_t)(n0 + j * 16 + lm) * D_ + kb + hi * 8);
      acc[j] = wmma16(a, b, acc[j]);
    }
  }

  if (z < 2) {
    _Float16* Out = (z == 0) ? Qh : Kh;
    const float smul = (z == 0) ? 0.125f : 1.0f;  // fold 1/sqrt(64) into Q
#pragma unroll
    for (int j = 0; j < 4; ++j) {
      const int n = n0 + j * 16 + lm;
      const float bval = bias[n];
#pragma unroll
      for (int r = 0; r < 8; ++r) {
        const int m = m0 + r + hi * 8;  // global row in [0, B*L)
        Out[(size_t)m * D_ + n] = (_Float16)((acc[j][r] + bval) * smul);
      }
    }
  } else {
#pragma unroll
    for (int j = 0; j < 4; ++j) {
      const int n = n0 + j * 16 + lm;
      const float bval = bias[n];
      const int hh = n >> 6, e = n & 63;
#pragma unroll
      for (int r = 0; r < 8; ++r) {
        const int m = m0 + r + hi * 8;
        const int bb = m >> 11, s = m & (S_ - 1);
        Vt[(((size_t)bb * H_ + hh) * E_ + e) * S_ + s] = (_Float16)(acc[j][r] + bval);
      }
    }
  }
}

// ---------------------------------------------------------------------------
// Kernel 2: flash attention. Block = 8 waves x 16 Q-rows. S-tiles of 64.
// Softmax denominator is accumulated on the matrix unit as a 5th accumulator
// (P @ ones), so only the row-max needs cross-lane shuffles.
// ---------------------------------------------------------------------------
__global__ void __launch_bounds__(256)
attn_kernel(const _Float16* __restrict__ Qh, const _Float16* __restrict__ Kh,
            const _Float16* __restrict__ Vt, _Float16* __restrict__ Oh) {
  __shared__ __align__(16) _Float16 Plds[8][16][72];  // per-wave P staging, padded pitch

  const int lane = threadIdx.x & 31;
  const int wave = threadIdx.x >> 5;
  const int lm = lane & 15;
  const int hi = lane >> 4;
  const int b = blockIdx.z;
  const int h = blockIdx.y;
  const int l0 = blockIdx.x * 128 + wave * 16;

  // Persistent Q A-fragments for E=64 (two K=32 steps); Q already pre-scaled.
  const _Float16* qrow = Qh + ((size_t)b * L_ + l0 + lm) * D_ + h * E_;
  const v16h qa0 = load_a_f16(qrow + 0 + hi * 8);
  const v16h qa1 = load_a_f16(qrow + 32 + hi * 8);

  // all-ones B fragment: accumulates row-sums of P (softmax denominator)
  v16h ones;
#pragma unroll
  for (int i = 0; i < 16; ++i) ones[i] = (_Float16)1.0f;

  v8f o[5];  // o[0..3] = P@V tile, o[4] = P@1 (denominator, replicated per col)
  float mrow[8];
#pragma unroll
  for (int j = 0; j < 5; ++j)
#pragma unroll
    for (int i = 0; i < 8; ++i) o[j][i] = 0.0f;
#pragma unroll
  for (int r = 0; r < 8; ++r) mrow[r] = -3.0e38f;

  const _Float16* kbase = Kh + (size_t)b * S_ * D_ + h * E_;
  const _Float16* vbase = Vt + ((size_t)b * H_ + h) * E_ * (size_t)S_;

  for (int st = 0; st < S_; st += 64) {
    // prefetch next K/V tiles (global_prefetch_b8) while this tile computes
    if (st + 64 < S_) {
      const _Float16* kn = kbase + (size_t)(st + 64 + lane) * D_;
      __builtin_prefetch(kn, 0, 3);
      __builtin_prefetch(kn + (size_t)32 * D_, 0, 3);
      const _Float16* vn = vbase + (size_t)lane * S_ + st + 64;
      __builtin_prefetch(vn, 0, 3);
      __builtin_prefetch(vn + (size_t)32 * S_, 0, 3);
    }

    // ---- scores: 16 x 64 tile, four 16x16 n-frags, K=64 via 2 wmma each ----
    v8f sc[4];
#pragma unroll
    for (int nf = 0; nf < 4; ++nf) {
#pragma unroll
      for (int i = 0; i < 8; ++i) sc[nf][i] = 0.0f;
      const _Float16* krow = kbase + (size_t)(st + nf * 16 + lm) * D_;
      sc[nf] = wmma16(qa0, load_b_f16(krow + hi * 8), sc[nf]);
      sc[nf] = wmma16(qa1, load_b_f16(krow + 32 + hi * 8), sc[nf]);
    }

    // ---- online softmax (row = r + 8*hi; 64 cols = 16 lanes x 4 frags) ----
#pragma unroll
    for (int r = 0; r < 8; ++r) {
      const float x0 = sc[0][r], x1 = sc[1][r], x2 = sc[2][r], x3 = sc[3][r];
      float mx = fmaxf(fmaxf(x0, x1), fmaxf(x2, x3));
      mx = fmaxf(mx, __shfl_xor(mx, 1));
      mx = fmaxf(mx, __shfl_xor(mx, 2));
      mx = fmaxf(mx, __shfl_xor(mx, 4));
      mx = fmaxf(mx, __shfl_xor(mx, 8));
      const float mnew = fmaxf(mrow[r], mx);
      const float corr = __expf(mrow[r] - mnew);
      mrow[r] = mnew;
      const float p0 = __expf(x0 - mnew);
      const float p1 = __expf(x1 - mnew);
      const float p2 = __expf(x2 - mnew);
      const float p3 = __expf(x3 - mnew);
#pragma unroll
      for (int j = 0; j < 5; ++j) o[j][r] *= corr;  // rescale P@V and P@1 alike
      // stage P (transpose rows<->lanes via LDS; same-wave LDS is in-order)
      _Float16* prow = &Plds[wave][r + hi * 8][lm];
      prow[0]  = (_Float16)p0;
      prow[16] = (_Float16)p1;
      prow[32] = (_Float16)p2;
      prow[48] = (_Float16)p3;
    }

    // ---- P (16x64) as two A-frags: accumulate P @ V and P @ ones ----
    const v16h pa0 = load_a_f16(&Plds[wave][lm][0 + hi * 8]);
    const v16h pa1 = load_a_f16(&Plds[wave][lm][32 + hi * 8]);
#pragma unroll
    for (int j = 0; j < 4; ++j) {
      const _Float16* vrow = vbase + (size_t)(j * 16 + lm) * S_ + st;
      o[j] = wmma16(pa0, load_b_f16(vrow + hi * 8), o[j]);
      o[j] = wmma16(pa1, load_b_f16(vrow + 32 + hi * 8), o[j]);
    }
    o[4] = wmma16(pa0, ones, o[4]);
    o[4] = wmma16(pa1, ones, o[4]);
  }

  // ---- finalize: divide by softmax denom, store f16 [B, L, 512] ----
#pragma unroll
  for (int r = 0; r < 8; ++r) {
    const float inv = 1.0f / o[4][r];
    const int l = l0 + r + hi * 8;
#pragma unroll
    for (int j = 0; j < 4; ++j) {
      const int n = h * E_ + j * 16 + lm;
      Oh[((size_t)b * L_ + l) * D_ + n] = (_Float16)(o[j][r] * inv);
    }
  }
}

// ---------------------------------------------------------------------------
// Kernel 3: output projection O(f16) @ Wo^T(f16) + bo -> f32 out
// ---------------------------------------------------------------------------
__global__ void __launch_bounds__(256)
out_proj_kernel(const _Float16* __restrict__ Oh, const _Float16* __restrict__ Wot,
                const float* __restrict__ bo, float* __restrict__ out) {
  const int lane = threadIdx.x & 31;
  const int wave = threadIdx.x >> 5;
  const int lm = lane & 15;
  const int hi = lane >> 4;
  const int m0 = blockIdx.x * 128 + wave * 16;
  const int n0 = blockIdx.y * 64;

  v8f acc[4];
#pragma unroll
  for (int j = 0; j < 4; ++j)
#pragma unroll
    for (int i = 0; i < 8; ++i) acc[j][i] = 0.0f;

  const _Float16* arow = Oh + (size_t)(m0 + lm) * D_;
  for (int kb = 0; kb < D_; kb += 32) {
    v16h a = load_a_f16(arow + kb + hi * 8);
#pragma unroll
    for (int j = 0; j < 4; ++j) {
      v16h b = load_b_f16(Wot + (size_t)(n0 + j * 16 + lm) * D_ + kb + hi * 8);
      acc[j] = wmma16(a, b, acc[j]);
    }
  }

#pragma unroll
  for (int j = 0; j < 4; ++j) {
    const int n = n0 + j * 16 + lm;
    const float bval = bo[n];
#pragma unroll
    for (int r = 0; r < 8; ++r) {
      const int m = m0 + r + hi * 8;
      out[(size_t)m * D_ + n] = acc[j][r] + bval;
    }
  }
}

// ---------------------------------------------------------------------------
extern "C" void kernel_launch(void* const* d_in, const int* in_sizes, int n_in,
                              void* d_out, int out_size, void* d_ws, size_t ws_size,
                              hipStream_t stream) {
  (void)in_sizes; (void)n_in; (void)out_size; (void)ws_size;
  const float* queries = (const float*)d_in[0];
  const float* keys    = (const float*)d_in[1];
  const float* values  = (const float*)d_in[2];
  const float* Wq = (const float*)d_in[3];
  const float* bq = (const float*)d_in[4];
  const float* Wk = (const float*)d_in[5];
  const float* bk = (const float*)d_in[6];
  const float* Wv = (const float*)d_in[7];
  const float* bv = (const float*)d_in[8];
  const float* Wo = (const float*)d_in[9];
  const float* bo = (const float*)d_in[10];
  float* out = (float*)d_out;

  char* ws = (char*)d_ws;
  _Float16* Wt = (_Float16*)(ws);                    // 4 x 512x512 f16 = 2 MB
  _Float16* Qh = (_Float16*)(ws + (2u << 20));       // [B,L,512] f16, 8 MB
  _Float16* Kh = (_Float16*)(ws + 10485760);         // [B,S,512] f16, 8 MB
  _Float16* Vt = (_Float16*)(ws + 18874368);         // [B,H,E,S] f16, 8 MB
  _Float16* Oh = (_Float16*)(ws + 27262976);         // [B,L,512] f16, 8 MB

  wconv_kernel<<<4096, 256, 0, stream>>>(Wq, Wk, Wv, Wo, Wt);
  qkv_proj_kernel<<<dim3(64, 8, 3), 256, 0, stream>>>(queries, keys, values, Wt,
                                                      bq, bk, bv, Qh, Kh, Vt);
  attn_kernel<<<dim3(16, 8, 4), 256, 0, stream>>>(Qh, Kh, Vt, Oh);
  out_proj_kernel<<<dim3(64, 8), 256, 0, stream>>>(Oh, Wt + 3 * (D_ * D_), bo, out);
}
